// LinkMLPPredictor_19198503813743
// MI455X (gfx1250) — matrix operand order repaired and checked by reading, not compiled, not run on hardware
//
#include <hip/hip_runtime.h>

typedef __attribute__((ext_vector_type(16))) _Float16 v16h;
typedef __attribute__((ext_vector_type(8)))  _Float16 v8h;
typedef __attribute__((ext_vector_type(4)))  _Float16 v4h;
typedef __attribute__((ext_vector_type(8)))  float    v8f;

#define INPUT_DIM 128
#define HIDDEN    128
#define KDIM      256     // concat(src,dst); tte column handled as rank-1 update
#define XS        264     // LDS row stride (halves): 132 dwords, mod 64 = 4 -> conflict-minimal
#define HS        136     // LDS row stride (halves): 68 dwords,  mod 64 = 4 -> conflict-minimal
#define MT        2       // M-tiles (of 16 edges) per block iteration
#define EPB       (16*MT) // edges per block iteration = 32

// ---------------------------------------------------------------------------
// Pack a fp32 weight matrix W[K][128] (row-major, k-major) into per-lane
// B-fragment order for v_wmma_f32_16x16x32_f16:
//   fragment idx = ((ntile*KT + ktile)*32 + lane), 16 contiguous halves/lane.
//   lane: n = ntile*16 + (lane&15); k = ktile*32 + (lane>=16 ? 8 : 0) + (h&7) + (h>=8 ? 16 : 0).
// ---------------------------------------------------------------------------
__global__ void pack_b_frag(const float* __restrict__ W, _Float16* __restrict__ Wp,
                            int KT) {
    int tid = blockIdx.x * blockDim.x + threadIdx.x;
    int total = 8 * KT * 32 * 16;            // NT = 128/16 = 8
    if (tid >= total) return;
    int h    = tid & 15;
    int lane = (tid >> 4) & 31;
    int g    = tid >> 9;
    int kt   = g % KT;
    int nt   = g / KT;
    int n = nt * 16 + (lane & 15);
    int k = kt * 32 + ((lane >> 4) << 3) + ((h >> 3) << 4) + (h & 7);
    Wp[tid] = (_Float16)W[k * HIDDEN + n];
}

// ---------------------------------------------------------------------------
// Fused edge-MLP: 32 edges (2 WMMA M-tiles) per block iteration, 8 waves,
// each wave owns one 16-wide N-tile of the 128 hidden units and loads its
// weight B-fragments from the packed buffers in every grid-stride iteration.
// ---------------------------------------------------------------------------
__global__ __launch_bounds__(256) void edge_mlp_kernel(
    const float* __restrict__ pat,  const float* __restrict__ cond,
    const int*   __restrict__ srcIdx, const int* __restrict__ dstIdx,
    const float* __restrict__ tte,
    const _Float16* __restrict__ W1p, const _Float16* __restrict__ Wrp,
    const _Float16* __restrict__ W2p,
    const float* __restrict__ W1, const float* __restrict__ Wr,
    const float* __restrict__ b1, const float* __restrict__ br,
    const float* __restrict__ b2, const float* __restrict__ Wo,
    const float* __restrict__ bo,
    float* __restrict__ out, int numTiles)
{
    __shared__ _Float16 xA [EPB * XS];
    __shared__ _Float16 h1A[EPB * HS];
    __shared__ int   sSrc[EPB];
    __shared__ int   sDst[EPB];
    __shared__ float sTte[EPB];
    __shared__ float sOut[EPB];

    const int tid   = threadIdx.x;
    const int wave  = tid >> 5;
    const int lane  = tid & 31;
    const int lh    = lane & 15;       // N offset within tile / A row index
    const int hg    = lane >> 4;       // half-group 0/1
    const int n     = wave * 16 + lh;  // hidden column this lane owns
    const int mbase = hg * 8;          // C-fragment row base (and A K-subbase)

    // Per-lane loop-invariant weight scalars
    const float w1l = W1[KDIM * HIDDEN + n];   // row 256 (tte coefficient)
    const float wrl = Wr[KDIM * HIDDEN + n];
    const float bb1 = b1[n];
    const float bbr = br[n];
    const float bb2 = b2[n];
    const float won = Wo[n];

    for (int tile = blockIdx.x; tile < numTiles; tile += gridDim.x) {
        const int e0 = tile * EPB;
        if (tid < EPB) {
            sSrc[tid] = srcIdx[e0 + tid];
            sDst[tid] = dstIdx[e0 + tid];
            sTte[tid] = tte[e0 + tid];
            sOut[tid] = 0.0f;
        }
        __syncthreads();

        // Gather 32 x 256 fp32 (L2-resident tables). Stage all 8 float4 loads
        // into registers first so they are all in flight, then cvt+store.
        float4 r[8];
        #pragma unroll
        for (int q = 0; q < 8; ++q) {
            int slot = tid + q * 256;            // 2048 slots
            int e = slot >> 6;
            int c = slot & 63;                   // float4 column 0..63
            const float4* rp = (c < 32)
                ? ((const float4*)(pat  + (size_t)sSrc[e] * INPUT_DIM) + c)
                : ((const float4*)(cond + (size_t)sDst[e] * INPUT_DIM) + (c - 32));
            r[q] = *rp;
        }
        #pragma unroll
        for (int q = 0; q < 8; ++q) {
            int slot = tid + q * 256;
            int e = slot >> 6;
            int c = slot & 63;
            v4h h4;
            h4[0] = (_Float16)r[q].x; h4[1] = (_Float16)r[q].y;
            h4[2] = (_Float16)r[q].z; h4[3] = (_Float16)r[q].w;
            *(v4h*)&xA[e * XS + c * 4] = h4;
        }
        __syncthreads();

        // ---- Layer 1: h1 = relu(x@W1 + b1 + tte*W1[256]); res = x@Wr + br
        v8f acc1[MT]; v8f accr[MT];
        #pragma unroll
        for (int m = 0; m < MT; ++m) { acc1[m] = (v8f){}; accr[m] = (v8f){}; }

        #pragma unroll
        for (int kt = 0; kt < 8; ++kt) {
            const int k0 = kt * 32 + mbase;          // A-frag K base for this lane
            v16h bf1 = ((const v16h*)W1p)[(wave * 8 + kt) * 32 + lane];
            v16h bfr = ((const v16h*)Wrp)[(wave * 8 + kt) * 32 + lane];
            #pragma unroll
            for (int m = 0; m < MT; ++m) {
                const int row = m * 16 + lh;
                v8h alo = *(const v8h*)&xA[row * XS + k0];
                v8h ahi = *(const v8h*)&xA[row * XS + k0 + 16];
                v16h a;
                #pragma unroll
                for (int i = 0; i < 8; ++i) { a[i] = alo[i]; a[i + 8] = ahi[i]; }
                acc1[m] = __builtin_amdgcn_wmma_f32_16x16x32_f16(
                              false, a, false, bf1, (short)0, acc1[m], false, false);
                accr[m] = __builtin_amdgcn_wmma_f32_16x16x32_f16(
                              false, a, false, bfr, (short)0, accr[m], false, false);
            }
        }

        #pragma unroll
        for (int m = 0; m < MT; ++m) {
            #pragma unroll
            for (int i = 0; i < 8; ++i) {
                const int row = m * 16 + mbase + i;
                float t   = sTte[row];
                float h1v = acc1[m][i] + t * w1l + bb1;
                h1v = h1v > 0.0f ? h1v : 0.0f;
                h1A[row * HS + n] = (_Float16)h1v;
                accr[m][i] += t * wrl + bbr;         // res (C seed for layer 2)
            }
        }
        __syncthreads();

        // ---- Layer 2: h2 = relu(h1@W2 + b2 + res)
        v8f acc2[MT];
        #pragma unroll
        for (int m = 0; m < MT; ++m) acc2[m] = accr[m];

        #pragma unroll
        for (int kt = 0; kt < 4; ++kt) {
            const int k0 = kt * 32 + mbase;
            v16h bf2 = ((const v16h*)W2p)[(wave * 4 + kt) * 32 + lane];
            #pragma unroll
            for (int m = 0; m < MT; ++m) {
                const int row = m * 16 + lh;
                v8h alo = *(const v8h*)&h1A[row * HS + k0];
                v8h ahi = *(const v8h*)&h1A[row * HS + k0 + 16];
                v16h a;
                #pragma unroll
                for (int i = 0; i < 8; ++i) { a[i] = alo[i]; a[i + 8] = ahi[i]; }
                acc2[m] = __builtin_amdgcn_wmma_f32_16x16x32_f16(
                              false, a, false, bf2, (short)0, acc2[m], false, false);
            }
        }

        // ---- Output: out[e] = h2 . Wo + bo  (reduce over this wave's 16 cols)
        #pragma unroll
        for (int m = 0; m < MT; ++m) {
            #pragma unroll
            for (int i = 0; i < 8; ++i) {
                float h2v = acc2[m][i] + bb2;
                h2v = h2v > 0.0f ? h2v : 0.0f;
                float v = h2v * won;
                v += __shfl_xor(v, 1, 16);
                v += __shfl_xor(v, 2, 16);
                v += __shfl_xor(v, 4, 16);
                v += __shfl_xor(v, 8, 16);
                if (lh == 0) atomicAdd(&sOut[m * 16 + mbase + i], v);
            }
        }
        __syncthreads();

        if (tid < EPB) out[e0 + tid] = sOut[tid] + bo[0];
        __syncthreads();   // protect shared buffers for next tile iteration
    }
}

// ---------------------------------------------------------------------------
extern "C" void kernel_launch(void* const* d_in, const int* in_sizes, int n_in,
                              void* d_out, int out_size, void* d_ws, size_t ws_size,
                              hipStream_t stream) {
    const float* pat  = (const float*)d_in[0];
    const float* cond = (const float*)d_in[1];
    const int*   eidx = (const int*)  d_in[2];
    const float* tte  = (const float*)d_in[3];
    const float* W1   = (const float*)d_in[4];
    const float* b1   = (const float*)d_in[5];
    const float* W2   = (const float*)d_in[6];
    const float* b2   = (const float*)d_in[7];
    const float* Wo   = (const float*)d_in[8];
    const float* bo   = (const float*)d_in[9];
    const float* Wr   = (const float*)d_in[10];
    const float* br   = (const float*)d_in[11];

    const int E        = in_sizes[3];       // tte length = N_EDGES
    const int numTiles = E / EPB;

    // Workspace: packed f16 B-fragments (W1: 64KB, Wr: 64KB, W2: 32KB)
    _Float16* W1p = (_Float16*)d_ws;
    _Float16* Wrp = W1p + 8 * 8 * 32 * 16;
    _Float16* W2p = Wrp + 8 * 8 * 32 * 16;

    pack_b_frag<<<(8 * 8 * 32 * 16 + 255) / 256, 256, 0, stream>>>(W1, W1p, 8);
    pack_b_frag<<<(8 * 8 * 32 * 16 + 255) / 256, 256, 0, stream>>>(Wr, Wrp, 8);
    pack_b_frag<<<(8 * 4 * 32 * 16 + 255) / 256, 256, 0, stream>>>(W2, W2p, 4);

    const int* srcIdx = eidx;
    const int* dstIdx = eidx + E;

    int grid = numTiles < 16384 ? numTiles : 16384;
    edge_mlp_kernel<<<grid, 256, 0, stream>>>(
        pat, cond, srcIdx, dstIdx, tte,
        W1p, Wrp, W2p,
        W1, Wr, b1, br, b2, Wo, bo,
        (float*)d_out, numTiles);
}
